// GATModel_16174846836807
// MI455X (gfx1250) — compile-verified, matrix operand order checked by reading, not run on hardware
//
#include <hip/hip_runtime.h>
#include <math.h>

// Problem constants (fixed by the reference)
constexpr int D  = 512;
constexpr int H  = 4;
constexpr int DH = 128;
constexpr float NEG_SLOPE = 0.2f;

typedef __attribute__((ext_vector_type(16))) _Float16 v16h;
typedef __attribute__((ext_vector_type(8)))  _Float16 v8h;
typedef __attribute__((ext_vector_type(8)))  float    v8f;

// ---------------------------------------------------------------------------
// Conversions / fills
// ---------------------------------------------------------------------------
__global__ void k_f32_to_f16(const float* __restrict__ in, _Float16* __restrict__ out, int n) {
    int id = blockIdx.x * blockDim.x + threadIdx.x;
    if (id < n) out[id] = (_Float16)in[id];
}

// W (H, D, DH) f32  ->  Wt[c][k] f16, c = h*DH+f (row-major by output column,
// contiguous in k) so B fragments load with the same pattern as A fragments.
__global__ void k_pack_W(const float* __restrict__ W, _Float16* __restrict__ Wt) {
    int id = blockIdx.x * blockDim.x + threadIdx.x;
    if (id >= H * D * DH) return;
    int h = id / (D * DH);
    int r = id % (D * DH);
    int d = r / DH;
    int f = r % DH;
    Wt[(size_t)(h * DH + f) * D + d] = (_Float16)W[id];
}

__global__ void k_fill(float* __restrict__ p, float v, int n) {
    int id = blockIdx.x * blockDim.x + threadIdx.x;
    if (id < n) p[id] = v;
}

// ---------------------------------------------------------------------------
// WMMA GEMM: C(M x 512) = A(M x 512, f16) * Bt^T (Bt is 512x512 f16, row = out col)
// One wave computes a 32x64 strip: 2 row tiles x 4 column tiles = 8 accumulators.
// The 4 B fragments are shared by both row tiles (2x arithmetic intensity per
// B load), K-loop step 32, software-pipelined (double-buffered fragments).
// Wave-uniform control flow -> EXEC all-1s at every WMMA (ISA requirement).
// M must be a multiple of 32 (N=20000 -> 625 row blocks).
// ---------------------------------------------------------------------------
__device__ __forceinline__ v16h ldfrag(const _Float16* __restrict__ p) {
    v8h lo = *(const v8h*)(p);
    v8h hi = *(const v8h*)(p + 16);
    return __builtin_shufflevector(lo, hi, 0,1,2,3,4,5,6,7,8,9,10,11,12,13,14,15);
}

__global__ __launch_bounds__(128) void k_gemm_wmma(const _Float16* __restrict__ A,
                                                   const _Float16* __restrict__ Bt,
                                                   float* __restrict__ C, int Mblocks) {
    const int wid  = blockIdx.x * (blockDim.x >> 5) + (threadIdx.x >> 5);
    const int lane = threadIdx.x & 31;
    const int mb = wid >> 3;          // 32-row block index
    const int cs = (wid & 7) * 64;    // 8 column strips of 64
    if (mb >= Mblocks) return;

    const int half = lane >> 4;       // K sub-chunk select per ISA A layout
    const int mr   = lane & 15;       // row (A) / column (B) within tile

    const _Float16* arow0 = A  + (size_t)(mb * 32 + mr)      * D + half * 8;
    const _Float16* arow1 = arow0 + (size_t)16 * D;           // second row tile
    const _Float16* bcol  = Bt + (size_t)(cs + mr)            * D + half * 8;
    const size_t bt_step  = (size_t)16 * D;                   // next 16-column tile

    v8f acc00 = {}; v8f acc01 = {}; v8f acc02 = {}; v8f acc03 = {};
    v8f acc10 = {}; v8f acc11 = {}; v8f acc12 = {}; v8f acc13 = {};

    // Prologue: fragments for k = 0
    v16h a0 = ldfrag(arow0);
    v16h a1 = ldfrag(arow1);
    v16h b0 = ldfrag(bcol);
    v16h b1 = ldfrag(bcol + bt_step);
    v16h b2 = ldfrag(bcol + 2 * bt_step);
    v16h b3 = ldfrag(bcol + 3 * bt_step);

    #pragma unroll
    for (int k0 = 0; k0 < D; k0 += 32) {
        // Prefetch next k-step into a distinct register set (dead on last iter)
        v16h a0n = a0, a1n = a1, b0n = b0, b1n = b1, b2n = b2, b3n = b3;
        const int kn = k0 + 32;
        if (kn < D) {
            a0n = ldfrag(arow0 + kn);
            a1n = ldfrag(arow1 + kn);
            b0n = ldfrag(bcol + kn);
            b1n = ldfrag(bcol + bt_step + kn);
            b2n = ldfrag(bcol + 2 * bt_step + kn);
            b3n = ldfrag(bcol + 3 * bt_step + kn);
        }

        acc00 = __builtin_amdgcn_wmma_f32_16x16x32_f16(false, a0, false, b0, (short)0, acc00, false, false);
        acc01 = __builtin_amdgcn_wmma_f32_16x16x32_f16(false, a0, false, b1, (short)0, acc01, false, false);
        acc02 = __builtin_amdgcn_wmma_f32_16x16x32_f16(false, a0, false, b2, (short)0, acc02, false, false);
        acc03 = __builtin_amdgcn_wmma_f32_16x16x32_f16(false, a0, false, b3, (short)0, acc03, false, false);
        acc10 = __builtin_amdgcn_wmma_f32_16x16x32_f16(false, a1, false, b0, (short)0, acc10, false, false);
        acc11 = __builtin_amdgcn_wmma_f32_16x16x32_f16(false, a1, false, b1, (short)0, acc11, false, false);
        acc12 = __builtin_amdgcn_wmma_f32_16x16x32_f16(false, a1, false, b2, (short)0, acc12, false, false);
        acc13 = __builtin_amdgcn_wmma_f32_16x16x32_f16(false, a1, false, b3, (short)0, acc13, false, false);

        a0 = a0n; a1 = a1n; b0 = b0n; b1 = b1n; b2 = b2n; b3 = b3n;
    }

    // C/D f32 layout: VGPR v -> row half*8+v, lanes 0..15 -> column mr
    #pragma unroll
    for (int v = 0; v < 8; ++v) {
        float* cp0 = C + (size_t)(mb * 32 + half * 8 + v)      * D + cs + mr;
        float* cp1 = C + (size_t)(mb * 32 + 16 + half * 8 + v) * D + cs + mr;
        cp0[0]  = acc00[v];
        cp0[16] = acc01[v];
        cp0[32] = acc02[v];
        cp0[48] = acc03[v];
        cp1[0]  = acc10[v];
        cp1[16] = acc11[v];
        cp1[32] = acc12[v];
        cp1[48] = acc13[v];
    }
}

// ---------------------------------------------------------------------------
// Per-node attention logits: als[n,h] = h[n,h,:].a_src[h], ald similarly
// ---------------------------------------------------------------------------
__global__ void k_node_alpha(const float* __restrict__ hfeat, const float* __restrict__ a,
                             float* __restrict__ als, float* __restrict__ ald, int NH) {
    int id = blockIdx.x * blockDim.x + threadIdx.x;
    if (id >= NH) return;
    int n = id / H, h = id % H;
    const float4* hp = (const float4*)(hfeat + (size_t)n * D + h * DH);
    const float4* as = (const float4*)(a + (size_t)h * 2 * DH);
    const float4* ad = (const float4*)(a + (size_t)h * 2 * DH + DH);
    float ss = 0.f, sd = 0.f;
    #pragma unroll 4
    for (int i = 0; i < DH / 4; ++i) {
        float4 hv = hp[i], av = as[i], dv = ad[i];
        ss += hv.x * av.x + hv.y * av.y + hv.z * av.z + hv.w * av.w;
        sd += hv.x * dv.x + hv.y * dv.y + hv.z * dv.z + hv.w * dv.w;
    }
    als[id] = ss;
    ald[id] = sd;
}

// float atomic max via int/uint ordering trick (init must be -inf)
__device__ __forceinline__ void atomicMaxFloat(float* addr, float val) {
    if (val >= 0.f) atomicMax((int*)addr, __float_as_int(val));
    else            atomicMin((unsigned int*)addr, __float_as_uint(val));
}

// Pass 1: e = leaky_relu(als[src]+ald[dst]); segment max into m[dst]
__global__ void k_edge1(const int* __restrict__ src, const int* __restrict__ dst,
                        const float* __restrict__ als, const float* __restrict__ ald,
                        float* __restrict__ ebuf, float* __restrict__ m, int EH) {
    int id = blockIdx.x * blockDim.x + threadIdx.x;
    if (id >= EH) return;
    int e = id / H, h = id % H;
    int sN = src[e], dN = dst[e];
    float v = als[sN * H + h] + ald[dN * H + h];
    v = v > 0.f ? v : v * NEG_SLOPE;
    ebuf[id] = v;
    atomicMaxFloat(&m[dN * H + h], v);
}

// Pass 2: ex = exp(e - m[dst]); segment sum into s[dst]
__global__ void k_edge2(const int* __restrict__ dst, float* __restrict__ ebuf,
                        const float* __restrict__ m, float* __restrict__ ssum, int EH) {
    int id = blockIdx.x * blockDim.x + threadIdx.x;
    if (id >= EH) return;
    int e = id / H, h = id % H;
    int dN = dst[e];
    float ex = expf(ebuf[id] - m[dN * H + h]);
    ebuf[id] = ex;
    atomicAdd(&ssum[dN * H + h], ex);
}

// Pass 3 (bandwidth-dominant): acc[dst,h,:] += (ex/s[dst,h]) * h[src,h,:]
// One 128-thread block per edge: thread t -> head t/32, 4 features via float4.
__global__ void k_edge3(const int* __restrict__ src, const int* __restrict__ dst,
                        const float* __restrict__ ebuf, const float* __restrict__ ssum,
                        const float* __restrict__ hfeat, float* __restrict__ acc) {
    int e   = blockIdx.x;
    int h   = threadIdx.x >> 5;
    int f0  = (threadIdx.x & 31) * 4;
    int sN = src[e], dN = dst[e];
    float alpha = ebuf[(size_t)e * H + h] / (ssum[dN * H + h] + 1e-16f);
    float4 hv = *(const float4*)(hfeat + (size_t)sN * D + h * DH + f0);
    float* ap = acc + (size_t)dN * D + h * DH + f0;
    atomicAdd(ap + 0, alpha * hv.x);
    atomicAdd(ap + 1, alpha * hv.y);
    atomicAdd(ap + 2, alpha * hv.z);
    atomicAdd(ap + 3, alpha * hv.w);
}

// xout = xin + elu(acc + b)
__global__ void k_finalize(const float* __restrict__ xin, const float* __restrict__ acc,
                           const float* __restrict__ b, float* __restrict__ xout, int ND) {
    int id = blockIdx.x * blockDim.x + threadIdx.x;
    if (id >= ND) return;
    float v = acc[id] + b[id % D];
    v = v > 0.f ? v : (expf(v) - 1.0f);
    xout[id] = xin[id] + v;
}

// ---------------------------------------------------------------------------
extern "C" void kernel_launch(void* const* d_in, const int* in_sizes, int n_in,
                              void* d_out, int out_size, void* d_ws, size_t ws_size,
                              hipStream_t stream) {
    const float* x  = (const float*)d_in[0];
    const int*   ei = (const int*)d_in[1];
    const float* W1 = (const float*)d_in[5];
    const float* a1 = (const float*)d_in[6];
    const float* b1 = (const float*)d_in[7];
    const float* W2 = (const float*)d_in[8];
    const float* a2 = (const float*)d_in[9];
    const float* b2 = (const float*)d_in[10];

    const int Nn = in_sizes[0] / D;     // 20000
    const int E  = in_sizes[1] / 2;     // 320000
    const int*   src = ei;
    const int*   dst = ei + E;

    // Carve workspace (256B aligned)
    char* base = (char*)d_ws;
    size_t off = 0;
    auto carve = [&](size_t bytes) -> void* {
        void* p = base + off;
        off = (off + bytes + 255) & ~(size_t)255;
        return p;
    };
    _Float16* xh    = (_Float16*)carve((size_t)Nn * D * sizeof(_Float16));
    _Float16* Wt    = (_Float16*)carve((size_t)D * D * sizeof(_Float16));
    float*    hfeat = (float*)carve((size_t)Nn * D * sizeof(float));
    float*    als   = (float*)carve((size_t)Nn * H * sizeof(float));
    float*    ald   = (float*)carve((size_t)Nn * H * sizeof(float));
    float*    mbuf  = (float*)carve((size_t)Nn * H * sizeof(float));
    float*    sbuf  = (float*)carve((size_t)Nn * H * sizeof(float));
    float*    ebuf  = (float*)carve((size_t)E * H * sizeof(float));
    float*    accb  = (float*)carve((size_t)Nn * D * sizeof(float));
    float*    x1    = (float*)carve((size_t)Nn * D * sizeof(float));
    (void)ws_size; (void)n_in; (void)out_size;

    const int ND = Nn * D;
    const int NH = Nn * H;
    const int EH = E * H;
    const int Mblocks = Nn / 32;         // 20000 % 32 == 0 -> 625
    const int waves   = Mblocks * 8;     // 8 column strips of 64

    auto run_layer = [&](const float* xin, const float* W, const float* avec,
                         const float* bvec, float* xout) {
        k_f32_to_f16<<<(ND + 255) / 256, 256, 0, stream>>>(xin, xh, ND);
        k_pack_W<<<(H * D * DH + 255) / 256, 256, 0, stream>>>(W, Wt);
        k_gemm_wmma<<<(waves + 3) / 4, 128, 0, stream>>>(xh, Wt, hfeat, Mblocks);
        k_node_alpha<<<(NH + 255) / 256, 256, 0, stream>>>(hfeat, avec, als, ald, NH);
        k_fill<<<(NH + 255) / 256, 256, 0, stream>>>(mbuf, -INFINITY, NH);
        k_fill<<<(NH + 255) / 256, 256, 0, stream>>>(sbuf, 0.f, NH);
        k_edge1<<<(EH + 255) / 256, 256, 0, stream>>>(src, dst, als, ald, ebuf, mbuf, EH);
        k_edge2<<<(EH + 255) / 256, 256, 0, stream>>>(dst, ebuf, mbuf, sbuf, EH);
        k_fill<<<(ND + 255) / 256, 256, 0, stream>>>(accb, 0.f, ND);
        k_edge3<<<E, 128, 0, stream>>>(src, dst, ebuf, sbuf, hfeat, accb);
        k_finalize<<<(ND + 255) / 256, 256, 0, stream>>>(xin, accb, bvec, xout, ND);
    };

    run_layer(x,  W1, a1, b1, x1);
    run_layer(x1, W2, a2, b2, (float*)d_out);
}
